// DecomposedAttention_28896539967854
// MI455X (gfx1250) — matrix operand order repaired and checked by reading, not compiled, not run on hardware
//
#include <hip/hip_runtime.h>

// ---------------------------------------------------------------------------
// Types
// ---------------------------------------------------------------------------
typedef __bf16 bf16_t;
typedef __attribute__((ext_vector_type(16))) bf16_t v16bf;
typedef __attribute__((ext_vector_type(8)))  bf16_t v8bf;
typedef __attribute__((ext_vector_type(4)))  bf16_t v4bf;
typedef __attribute__((ext_vector_type(8)))  float  v8f;

static __device__ __forceinline__ bf16_t f2bf(float f) {
    union { float f; unsigned u; } c; c.f = f;
    unsigned r = c.u + 0x7FFFu + ((c.u >> 16) & 1u);   // round-to-nearest-even
    union { unsigned short s; bf16_t b; } o; o.s = (unsigned short)(r >> 16);
    return o.b;
}
static __device__ __forceinline__ float bf2f(bf16_t b) {
    union { unsigned short s; bf16_t b; } i; i.b = b;
    union { unsigned u; float f; } o; o.u = ((unsigned)i.s) << 16;
    return o.f;
}

// CDNA5 async global->LDS copy (16B per lane), tracked by ASYNCcnt.
// LDS destination = low 32 bits of generic shared pointer (aperture is in
// the upper 32 bits per the ISA aperture check).
static __device__ __forceinline__ void async_ld_b128(void* lds_ptr, const void* gaddr) {
    unsigned lds = (unsigned)(size_t)lds_ptr;
    asm volatile("global_load_async_to_lds_b128 %0, %1, off"
                 :: "v"(lds), "v"(gaddr) : "memory");
}
static __device__ __forceinline__ void wait_async0() {
    asm volatile("s_wait_asynccnt 0" ::: "memory");
}

// A fragment, 16x32 bf16.  ISA layout: lane l holds row (l&15);
// VGPR0..3 = k in [kg, kg+8), VGPR4..7 = k in [kg+16, kg+24), kg=(l>>4)*8.
static __device__ __forceinline__ v16bf frag_a(const bf16_t* base, int ldk, int lane) {
    int r  = lane & 15;
    int kg = (lane >> 4) << 3;
    const bf16_t* p = base + r * ldk + kg;
    union { v16bf v; v8bf h[2]; } u;
    u.h[0] = *(const v8bf*)(p);
    u.h[1] = *(const v8bf*)(p + 16);
    return u.v;
}
// B fragment, 32x16 bf16, LDS stored [N][K]: lane l holds column (l&15),
// k in [kb, kb+8) then [kb+16, kb+24), kb=(l>>4)*8 (same order as A).
static __device__ __forceinline__ v16bf frag_b(const bf16_t* base, int ldk, int lane) {
    int n  = lane & 15;
    int kb = (lane >> 4) << 3;
    const bf16_t* p = base + n * ldk + kb;
    union { v16bf v; v8bf h[2]; } u;
    u.h[0] = *(const v8bf*)(p);
    u.h[1] = *(const v8bf*)(p + 16);
    return u.v;
}
static __device__ __forceinline__ v8f wmma_bf16(v16bf a, v16bf b, v8f c) {
    return __builtin_amdgcn_wmma_f32_16x16x32_bf16(false, a, false, b,
                                                   (short)0, c, false, false);
}

// ---------------------------------------------------------------------------
// Prep: fp32 [K][N] -> bf16 [N][K] (transpose + convert)
// ---------------------------------------------------------------------------
__global__ void wt_cvt_kernel(const float* __restrict__ in, bf16_t* __restrict__ out,
                              int K, int N) {
    int idx = blockIdx.x * blockDim.x + threadIdx.x;
    if (idx >= K * N) return;
    int n = idx / K, k = idx % K;
    out[idx] = f2bf(in[(size_t)k * N + n]);
}

// Prep: fp32 -> bf16 elementwise (x), 4 elems/thread
__global__ void x_cvt_kernel(const float* __restrict__ in, bf16_t* __restrict__ out,
                             int n4) {
    int idx = blockIdx.x * blockDim.x + threadIdx.x;
    if (idx >= n4) return;
    float4 f = ((const float4*)in)[idx];
    v4bf pk;
    pk[0] = f2bf(f.x); pk[1] = f2bf(f.y); pk[2] = f2bf(f.z); pk[3] = f2bf(f.w);
    ((v4bf*)out)[idx] = pk;
}

// ---------------------------------------------------------------------------
// GEMM 1: qkv = bf16( xb @ w_qkv ),  A bf16 [M][K], B = Wt bf16 [N][K]
// 128x128 tile, BK=64, 256 threads (8 waves, 4x2), wave tile 32x64.
// Tiles staged with async global->LDS.
// ---------------------------------------------------------------------------
#define BM 128
#define BN 128
#define BK 64

__global__ void __launch_bounds__(256)
qkv_gemm_kernel(const bf16_t* __restrict__ Xb, const bf16_t* __restrict__ Wt,
                bf16_t* __restrict__ Out, int M, int N, int K) {
    __shared__ bf16_t As[BM * BK];   // [m][k] 16KB
    __shared__ bf16_t Bs[BN * BK];   // [n][k] 16KB
    const int tid   = threadIdx.x;
    const int lane  = tid & 31;
    const int wave  = tid >> 5;
    const int waveM = wave >> 1;      // 0..3
    const int waveN = wave & 1;       // 0..1
    const int mBase = blockIdx.y * BM;
    const int nBase = blockIdx.x * BN;

    v8f acc[2][4];
#pragma unroll
    for (int i = 0; i < 2; ++i)
#pragma unroll
        for (int j = 0; j < 4; ++j) acc[i][j] = (v8f)0.0f;

    for (int kb = 0; kb < K; kb += BK) {
        // A + B tiles: 2x 128x64 bf16, 1024 chunks of 16B each, async to LDS
#pragma unroll
        for (int it = 0; it < 4; ++it) {
            int idx = it * 256 + tid;
            int row = idx >> 3;
            int c8  = idx & 7;
            async_ld_b128(As + row * BK + c8 * 8,
                          Xb + (size_t)(mBase + row) * K + kb + c8 * 8);
            async_ld_b128(Bs + row * BK + c8 * 8,
                          Wt + (size_t)(nBase + row) * K + kb + c8 * 8);
        }
        wait_async0();
        __syncthreads();

#pragma unroll
        for (int ks = 0; ks < BK; ks += 32) {
            v16bf af[2], bfg[4];
#pragma unroll
            for (int mt = 0; mt < 2; ++mt)
                af[mt] = frag_a(As + (waveM * 32 + mt * 16) * BK + ks, BK, lane);
#pragma unroll
            for (int nt = 0; nt < 4; ++nt)
                bfg[nt] = frag_b(Bs + (waveN * 64 + nt * 16) * BK + ks, BK, lane);
#pragma unroll
            for (int mt = 0; mt < 2; ++mt)
#pragma unroll
                for (int nt = 0; nt < 4; ++nt)
                    acc[mt][nt] = wmma_bf16(af[mt], bfg[nt], acc[mt][nt]);
        }
        __syncthreads();
    }

    const int halo = lane >> 4, ncol = lane & 15;
#pragma unroll
    for (int mt = 0; mt < 2; ++mt)
#pragma unroll
        for (int nt = 0; nt < 4; ++nt)
#pragma unroll
            for (int v = 0; v < 8; ++v) {
                int r = mBase + waveM * 32 + mt * 16 + halo * 8 + v;
                int c = nBase + waveN * 64 + nt * 16 + ncol;
                Out[(size_t)r * N + c] = f2bf(acc[mt][nt][v]);
            }
}

// ---------------------------------------------------------------------------
// Flash attention over one decomposition.
// row(s) = b*13824 + oi*groupStride + (s/24)*sOuter + (s%24)*sInner
// L=576, hd=64.  Block: 128 thr (4 waves), 64 q rows (16 per wave),
// kv tiles of 64.  Online softmax; 16x16x32 bf16 WMMA.
// Q/K tiles staged with async global->LDS; V transposed through VGPRs.
// ---------------------------------------------------------------------------
__global__ void __launch_bounds__(128)
attn_kernel(const bf16_t* __restrict__ QKV, bf16_t* __restrict__ Out,
            int groupStride, int sOuter, int sInner) {
    __shared__ bf16_t Qs[64 * 64];        // [q][d]
    __shared__ bf16_t Ks[64 * 64];        // [kv][d]
    __shared__ bf16_t Vs[64 * 64];        // transposed: [d][kv]
    __shared__ bf16_t Ps[4][16 * 64];     // per-wave P staging [q][kv]

    const int bx   = blockIdx.x;
    const int qb   = bx % 9;
    const int head = (bx / 9) % 8;
    const int g    = bx / 72;
    const long base = (long)(g / 24) * 13824 + (long)(g % 24) * groupStride;

    const int tid  = threadIdx.x;
    const int lane = tid & 31;
    const int wave = tid >> 5;
    const int LDQ  = 1536;

    const bf16_t* Qg = QKV + head * 64;
    const bf16_t* Kg = QKV + 512 + head * 64;
    const bf16_t* Vg = QKV + 1024 + head * 64;

    auto rowOf = [&](int s) -> long {
        return base + (long)(s / 24) * sOuter + (long)(s % 24) * sInner;
    };

    // Load 64x64 Q tile (async)
#pragma unroll
    for (int it = 0; it < 4; ++it) {
        int idx = it * 128 + tid;
        int r = idx >> 3, c = idx & 7;
        long gr = rowOf(qb * 64 + r);
        async_ld_b128(Qs + r * 64 + c * 8, Qg + gr * LDQ + c * 8);
    }

    float mstate[8], lstate[8];
    v8f acc[4];
#pragma unroll
    for (int v = 0; v < 8; ++v) { mstate[v] = -1e30f; lstate[v] = 0.0f; }
#pragma unroll
    for (int nt = 0; nt < 4; ++nt) acc[nt] = (v8f)0.0f;

    for (int kb = 0; kb < 9; ++kb) {
        __syncthreads();                       // previous iter done with Ks/Vs
        // K tile (async) + V tile (transposed via VGPRs)
#pragma unroll
        for (int it = 0; it < 4; ++it) {
            int idx = it * 128 + tid;
            int r = idx >> 3, c = idx & 7;
            long gr = rowOf(kb * 64 + r);
            async_ld_b128(Ks + r * 64 + c * 8, Kg + gr * LDQ + c * 8);
            v8bf vv = *(const v8bf*)(Vg + gr * LDQ + c * 8);
#pragma unroll
            for (int j = 0; j < 8; ++j) Vs[(c * 8 + j) * 64 + r] = vv[j];
        }
        wait_async0();                         // covers this wave's Q + K asyncs
        __syncthreads();

        // S = Q @ K^T (16x64 per wave)
        v8f st[4];
#pragma unroll
        for (int nt = 0; nt < 4; ++nt) st[nt] = (v8f)0.0f;
#pragma unroll
        for (int ks = 0; ks < 64; ks += 32) {
            v16bf aq = frag_a(Qs + wave * 16 * 64 + ks, 64, lane);
#pragma unroll
            for (int nt = 0; nt < 4; ++nt) {
                v16bf bk = frag_b(Ks + nt * 16 * 64 + ks, 64, lane);
                st[nt] = wmma_bf16(aq, bk, st[nt]);
            }
        }
        // reference multiplies q by sqrt(hd)=8
#pragma unroll
        for (int nt = 0; nt < 4; ++nt)
#pragma unroll
            for (int v = 0; v < 8; ++v) st[nt][v] = st[nt][v] * 8.0f;

        // Online softmax (row reductions across 16-lane halves)
        float mnew[8], corr[8], rsum[8];
#pragma unroll
        for (int v = 0; v < 8; ++v) {
            float mx = fmaxf(fmaxf(st[0][v], st[1][v]), fmaxf(st[2][v], st[3][v]));
#pragma unroll
            for (int x = 1; x < 16; x <<= 1) mx = fmaxf(mx, __shfl_xor(mx, x, 32));
            mnew[v] = fmaxf(mstate[v], mx);
            corr[v] = __expf(mstate[v] - mnew[v]);
            mstate[v] = mnew[v];
            rsum[v] = 0.0f;
        }
#pragma unroll
        for (int nt = 0; nt < 4; ++nt)
#pragma unroll
            for (int v = 0; v < 8; ++v) {
                float p = __expf(st[nt][v] - mnew[v]);
                st[nt][v] = p;
                rsum[v] += p;
            }
#pragma unroll
        for (int v = 0; v < 8; ++v) {
            float rs = rsum[v];
#pragma unroll
            for (int x = 1; x < 16; x <<= 1) rs += __shfl_xor(rs, x, 32);
            lstate[v] = lstate[v] * corr[v] + rs;
#pragma unroll
            for (int nt = 0; nt < 4; ++nt) acc[nt][v] = acc[nt][v] * corr[v];
        }

        // Stage P (C layout -> LDS row-major [q][kv]) for use as A fragment
        bf16_t* Pw = &Ps[wave][0];
        {
            const int halo = lane >> 4, ncol = lane & 15;
#pragma unroll
            for (int nt = 0; nt < 4; ++nt)
#pragma unroll
                for (int v = 0; v < 8; ++v)
                    Pw[(halo * 8 + v) * 64 + nt * 16 + ncol] = f2bf(st[nt][v]);
        }
        __syncthreads();

        // acc += P @ V   (B = Vs[d][kv])
#pragma unroll
        for (int ks = 0; ks < 64; ks += 32) {
            v16bf ap = frag_a(Pw + ks, 64, lane);
#pragma unroll
            for (int nt = 0; nt < 4; ++nt) {
                v16bf bv = frag_b(Vs + nt * 16 * 64 + ks, 64, lane);
                acc[nt] = wmma_bf16(ap, bv, acc[nt]);
            }
        }
    }

    // Epilogue: o = acc / l  -> bf16 at [row(s)][head*64 + d]
    {
        const int halo = lane >> 4, ncol = lane & 15;
#pragma unroll
        for (int v = 0; v < 8; ++v) {
            float inv = 1.0f / lstate[v];
            long gr = rowOf(qb * 64 + wave * 16 + halo * 8 + v);
#pragma unroll
            for (int nt = 0; nt < 4; ++nt)
                Out[gr * 512 + head * 64 + nt * 16 + ncol] = f2bf(acc[nt][v] * inv);
        }
    }
}

// ---------------------------------------------------------------------------
// GEMM 2: out(f32) = (o1+o2+o3) @ w_proj,  B = wprojT bf16 [N][K]
// A summed through VGPRs; B tile async.
// ---------------------------------------------------------------------------
__global__ void __launch_bounds__(256)
proj_gemm_kernel(const bf16_t* __restrict__ O1, const bf16_t* __restrict__ O2,
                 const bf16_t* __restrict__ O3, const bf16_t* __restrict__ Wt,
                 float* __restrict__ Out, int M, int N, int K) {
    __shared__ bf16_t As[BM * BK];
    __shared__ bf16_t Bs[BN * BK];
    const int tid   = threadIdx.x;
    const int lane  = tid & 31;
    const int wave  = tid >> 5;
    const int waveM = wave >> 1;
    const int waveN = wave & 1;
    const int mBase = blockIdx.y * BM;
    const int nBase = blockIdx.x * BN;

    v8f acc[2][4];
#pragma unroll
    for (int i = 0; i < 2; ++i)
#pragma unroll
        for (int j = 0; j < 4; ++j) acc[i][j] = (v8f)0.0f;

    for (int kb = 0; kb < K; kb += BK) {
#pragma unroll
        for (int it = 0; it < 4; ++it) {
            int idx = it * 256 + tid;
            int row = idx >> 3, c8 = idx & 7;
            size_t go = (size_t)(mBase + row) * K + kb + c8 * 8;
            async_ld_b128(Bs + row * BK + c8 * 8, Wt + (size_t)(nBase + row) * K + kb + c8 * 8);
            v8bf a = *(const v8bf*)(O1 + go);
            v8bf b = *(const v8bf*)(O2 + go);
            v8bf c = *(const v8bf*)(O3 + go);
            v8bf r;
#pragma unroll
            for (int j = 0; j < 8; ++j) r[j] = f2bf(bf2f(a[j]) + bf2f(b[j]) + bf2f(c[j]));
            *(v8bf*)(As + row * BK + c8 * 8) = r;
        }
        wait_async0();
        __syncthreads();
#pragma unroll
        for (int ks = 0; ks < BK; ks += 32) {
            v16bf af[2], bfg[4];
#pragma unroll
            for (int mt = 0; mt < 2; ++mt)
                af[mt] = frag_a(As + (waveM * 32 + mt * 16) * BK + ks, BK, lane);
#pragma unroll
            for (int nt = 0; nt < 4; ++nt)
                bfg[nt] = frag_b(Bs + (waveN * 64 + nt * 16) * BK + ks, BK, lane);
#pragma unroll
            for (int mt = 0; mt < 2; ++mt)
#pragma unroll
                for (int nt = 0; nt < 4; ++nt)
                    acc[mt][nt] = wmma_bf16(af[mt], bfg[nt], acc[mt][nt]);
        }
        __syncthreads();
    }

    const int halo = lane >> 4, ncol = lane & 15;
#pragma unroll
    for (int mt = 0; mt < 2; ++mt)
#pragma unroll
        for (int nt = 0; nt < 4; ++nt)
#pragma unroll
            for (int v = 0; v < 8; ++v) {
                int r = mBase + waveM * 32 + mt * 16 + halo * 8 + v;
                int c = nBase + waveN * 64 + nt * 16 + ncol;
                Out[(size_t)r * N + c] = acc[mt][nt][v];
            }
}

// ---------------------------------------------------------------------------
// Launch
// ---------------------------------------------------------------------------
extern "C" void kernel_launch(void* const* d_in, const int* in_sizes, int n_in,
                              void* d_out, int out_size, void* d_ws, size_t ws_size,
                              hipStream_t stream) {
    (void)in_sizes; (void)n_in; (void)out_size; (void)ws_size;
    const float* x      = (const float*)d_in[0];   // (2,13824,512)
    const float* w_qkv  = (const float*)d_in[1];   // (512,1536)
    const float* w_proj = (const float*)d_in[2];   // (512,512)
    float* out = (float*)d_out;

    const int M = 2 * 13824;     // 27648
    const int C = 512;
    const int N1 = 3 * C;        // 1536

    char* ws = (char*)d_ws;
    size_t off = 0;
    auto alloc = [&](size_t bytes) -> void* {
        off = (off + 255) & ~(size_t)255;
        void* p = ws + off;
        off += bytes;
        return p;
    };
    bf16_t* wqkvT  = (bf16_t*)alloc((size_t)N1 * C * 2);
    bf16_t* wprojT = (bf16_t*)alloc((size_t)C * C * 2);
    bf16_t* xb     = (bf16_t*)alloc((size_t)M * C * 2);
    bf16_t* qkv    = (bf16_t*)alloc((size_t)M * N1 * 2);
    bf16_t* o1     = (bf16_t*)alloc((size_t)M * C * 2);
    bf16_t* o2     = (bf16_t*)alloc((size_t)M * C * 2);
    bf16_t* o3     = (bf16_t*)alloc((size_t)M * C * 2);

    wt_cvt_kernel<<<(C * N1 + 255) / 256, 256, 0, stream>>>(w_qkv, wqkvT, C, N1);
    wt_cvt_kernel<<<(C * C + 255) / 256, 256, 0, stream>>>(w_proj, wprojT, C, C);
    {
        int n4 = M * C / 4;
        x_cvt_kernel<<<(n4 + 255) / 256, 256, 0, stream>>>(x, xb, n4);
    }

    qkv_gemm_kernel<<<dim3(N1 / BN, M / BM), 256, 0, stream>>>(xb, wqkvT, qkv, M, N1, C);

    // 9 q-blocks * 8 heads * 48 groups = 3456 blocks per decomposition
    const int nblk = 9 * 8 * 48;
    // dec1 (hw): group over t (stride 1),  s=h*24+w  -> outer 576, inner 24
    attn_kernel<<<nblk, 128, 0, stream>>>(qkv, o1, 1, 576, 24);
    // dec2 (ht): group over w (stride 24), s=h*24+t  -> outer 576, inner 1
    attn_kernel<<<nblk, 128, 0, stream>>>(qkv, o2, 24, 576, 1);
    // dec3 (wt): group over h (stride 576), s=w*24+t -> outer 24, inner 1
    attn_kernel<<<nblk, 128, 0, stream>>>(qkv, o3, 576, 24, 1);

    proj_gemm_kernel<<<dim3(C / BN, M / BM), 256, 0, stream>>>(o1, o2, o3, wprojT, out, M, C, C);
}
